// SpectreLinear_29334626632015
// MI455X (gfx1250) — matrix operand order repaired and checked by reading, not compiled, not run on hardware
//
#include <hip/hip_runtime.h>
#include <hip/hip_bf16.h>

typedef __attribute__((ext_vector_type(16))) __bf16 v16bf;
typedef __attribute__((ext_vector_type(8)))  float  v8f;

#define B_          32
#define N_          2048
#define E_          768
#define OUT_        384
#define HID_        16
#define LOCALC_     192
#define ROWS_TOTAL  (B_ * N_)          // 65536
#define ROWS_PER_WG 64
#define NGROUPS     (ROWS_TOTAL / ROWS_PER_WG)  // 1024
#define NBLOCKS     256
#define NTHREADS    128

// ---- LDS layout (bytes). Phase-1 region (gW + Gtile) overlaps phase-2+ region (lw + Y).
#define OFF_GFEAT 0                         // 64 f32
#define OFF_MU    256                       // 64 f32
#define OFF_RS    512                       // 64 f32
#define OFF_GW_HI 1024                      // 16*768 bf16 (24576 B)
#define OFF_GW_LO (OFF_GW_HI + 16*768*2)    // 25600
#define OFF_GTILE (OFF_GW_LO + 16*768*2)    // 50176, 64*16 f32
#define OFF_LW_HI 1024                      // 16*192 bf16 (6144 B)
#define OFF_LW_LO (OFF_LW_HI + 16*192*2)    // 7168
#define OFF_Y     (OFF_LW_LO + 16*192*2)    // 13312, 64*384 f32
#define SMEM_BYTES (OFF_Y + ROWS_PER_WG*OUT_*4)  // 111616

// ---- workspace layout (pre-split weights, bf16 hi/lo), sizes in uint32 units
#define WS_GW_U    6144                      // 16*768/2
#define WS_LW_U    36864                     // 192*384/2
#define WS_BYTES   ((WS_GW_U*2 + WS_LW_U*2) * 4)   // 344064
#define WS_LWHI_B  (WS_GW_U * 2 * 4)         // 49152 byte offset of lw-hi
#define WS_LWLO_B  (WS_LWHI_B + WS_LW_U * 4) // 196608

union BfOp {
  v16bf v;
  unsigned u[8];
  uint4 q[2];
};

// Split a pair of f32 into packed bf16 hi (RNE) and bf16 lo (truncated correction).
__device__ __forceinline__ void split_pair(float x0, float x1, unsigned& hi, unsigned& lo) {
  unsigned u0 = __builtin_bit_cast(unsigned, x0);
  unsigned u1 = __builtin_bit_cast(unsigned, x1);
  unsigned r0 = u0 + 0x7FFFu + ((u0 >> 16) & 1u);
  unsigned r1 = u1 + 0x7FFFu + ((u1 >> 16) & 1u);
  hi = __builtin_amdgcn_perm(r1, r0, 0x07060302);        // {r1.hi16, r0.hi16}
  float f0 = __builtin_bit_cast(float, r0 & 0xFFFF0000u);
  float f1 = __builtin_bit_cast(float, r1 & 0xFFFF0000u);
  unsigned v0 = __builtin_bit_cast(unsigned, x0 - f0);
  unsigned v1 = __builtin_bit_cast(unsigned, x1 - f1);
  lo = __builtin_amdgcn_perm(v1, v0, 0x07060302);        // truncation is fine for the lo term
}

__device__ __forceinline__ float gelu_exact(float v) {
  return 0.5f * v * (1.0f + erff(v * 0.70710678118654752440f));
}

// Async copy of 16 bytes global -> LDS (per lane), tracked on ASYNCcnt.
__device__ __forceinline__ void async_copy16(unsigned lds_off, const void* gbase, unsigned goff) {
  asm volatile("global_load_async_to_lds_b128 %0, %1, %2"
               :: "v"(lds_off), "v"(goff), "s"(gbase) : "memory");
}
__device__ __forceinline__ void async_wait0() {
  asm volatile("s_wait_asynccnt 0" ::: "memory");
}

// ---------------- prep: pre-split weights into workspace ----------------
__global__ __launch_bounds__(256)
void spectre_prep(const float* __restrict__ w_g1, const float* __restrict__ w_l,
                  unsigned* __restrict__ ws)
{
  unsigned* gWhi = ws;
  unsigned* gWlo = ws + WS_GW_U;
  unsigned* lwhi = ws + 2 * WS_GW_U;
  unsigned* lwlo = ws + 2 * WS_GW_U + WS_LW_U;
  for (int idx = blockIdx.x * 256 + threadIdx.x; idx < WS_GW_U + WS_LW_U;
       idx += gridDim.x * 256) {
    if (idx < WS_GW_U) {
      // w_g1^T, layout [n][k] k-fastest, pairs along k
      int n = idx / 384, kp = idx - n * 384, k = 2 * kp;
      split_pair(w_g1[k * HID_ + n], w_g1[k * HID_ + HID_ + n], gWhi[idx], gWlo[idx]);
    } else {
      // w_l, layout [nt][n][k] k-fastest, pairs along k
      int j = idx - WS_GW_U;
      int nt = j / 1536, r = j - nt * 1536, n = r / 96, kp = r - n * 96, k = 2 * kp;
      int src = k * OUT_ + nt * 16 + n;
      split_pair(w_l[src], w_l[src + OUT_], lwhi[j], lwlo[j]);
    }
  }
}

// ---------------- fused main kernel ----------------
__global__ __launch_bounds__(NTHREADS, 1)
void spectre_fused(const float* __restrict__ x,
                   const float* __restrict__ w_g1,  const float* __restrict__ b_g1,
                   const float* __restrict__ ln_g_w, const float* __restrict__ ln_g_b,
                   const float* __restrict__ w_g2,  const float* __restrict__ b_g2,
                   const float* __restrict__ w_l,   const float* __restrict__ b_l,
                   const float* __restrict__ ln_l_w, const float* __restrict__ ln_l_b,
                   const unsigned* __restrict__ ws,   // pre-split weights or nullptr
                   float* __restrict__ out)
{
  __shared__ __align__(16) char smem[SMEM_BYTES];
  float* gfeat = (float*)(smem + OFF_GFEAT);
  float* muA   = (float*)(smem + OFF_MU);
  float* rsA   = (float*)(smem + OFF_RS);
  float* Gtile = (float*)(smem + OFF_GTILE);
  float* Ytile = (float*)(smem + OFF_Y);

  const int tid  = threadIdx.x;
  const int lane = tid & 31;
  const int wv   = tid >> 5;
  const int ml   = lane & 15;   // row (A) or col (B/C) index within 16
  const int hh   = lane >> 4;   // half selector for WMMA lane layout
  const char* wsb = (const char*)ws;

  for (int g = blockIdx.x; g < NGROUPS; g += gridDim.x) {
    __syncthreads();  // protect LDS reuse across group iterations

    // ---------- stage w_g1^T (bf16 hi/lo, [n][k] k-fastest) ----------
    if (ws) {
      // flat async copy: 49152 B (hi then lo, contiguous in both ws and LDS)
      for (int i = tid * 16; i < 16 * 768 * 2 * 2; i += NTHREADS * 16)
        async_copy16(OFF_GW_HI + i, wsb, i);
      async_wait0();
    } else {
      unsigned* dsthi = (unsigned*)(smem + OFF_GW_HI);
      unsigned* dstlo = (unsigned*)(smem + OFF_GW_LO);
      for (int i = tid; i < 16 * 384; i += NTHREADS) {
        int n = i / 384, kp = i - n * 384, k = 2 * kp;
        split_pair(w_g1[k * HID_ + n], w_g1[k * HID_ + HID_ + n], dsthi[i], dstlo[i]);
      }
    }
    __syncthreads();

    const int row0 = g * ROWS_PER_WG + wv * 16;
    const float* xrow = x + (size_t)(row0 + ml) * E_;

    // ---------- phase 1: global branch (16x768)@(768x16), bf16x3 WMMA ----------
    v8f cg = {};
    for (int kt = 0; kt < 24; ++kt) {
      // A tile: K=16h+c (contiguous order, same as B)
      const float* ap = xrow + kt * 32 + 16 * hh;
      float4 a0 = *(const float4*)(ap);
      float4 a1 = *(const float4*)(ap + 4);
      float4 a2 = *(const float4*)(ap + 8);
      float4 a3 = *(const float4*)(ap + 12);
      float av[16] = {a0.x, a0.y, a0.z, a0.w, a1.x, a1.y, a1.z, a1.w,
                      a2.x, a2.y, a2.z, a2.w, a3.x, a3.y, a3.z, a3.w};
      BfOp ah, al;
      #pragma unroll
      for (int c2 = 0; c2 < 8; ++c2)
        split_pair(av[2 * c2], av[2 * c2 + 1], ah.u[c2], al.u[c2]);

      // B tile: lane n=ml holds K = kt*32 + 16*hh + c (16 contiguous bf16)
      BfOp bh, bl;
      const uint4* bph = (const uint4*)(smem + OFF_GW_HI + (size_t)(ml * 768 + kt * 32 + 16 * hh) * 2);
      bh.q[0] = bph[0]; bh.q[1] = bph[1];
      const uint4* bpl = (const uint4*)(smem + OFF_GW_LO + (size_t)(ml * 768 + kt * 32 + 16 * hh) * 2);
      bl.q[0] = bpl[0]; bl.q[1] = bpl[1];

      cg = __builtin_amdgcn_wmma_f32_16x16x32_bf16(false, ah.v, false, bh.v, (short)0, cg, false, false);
      cg = __builtin_amdgcn_wmma_f32_16x16x32_bf16(false, al.v, false, bh.v, (short)0, cg, false, false);
      cg = __builtin_amdgcn_wmma_f32_16x16x32_bf16(false, ah.v, false, bl.v, (short)0, cg, false, false);
    }
    {
      float bg = b_g1[ml];
      #pragma unroll
      for (int r = 0; r < 8; ++r)
        Gtile[(wv * 16 + r + 8 * hh) * 16 + ml] = cg[r] + bg;  // C: VGPR r -> M=r+8h, N=lane
    }
    __syncthreads();

    // per-row LN(16) + GELU + dot(w_g2) + b_g2
    if (lane < 16) {
      const float* gr = Gtile + (wv * 16 + lane) * 16;
      float s = 0.f, q = 0.f;
      #pragma unroll
      for (int j = 0; j < 16; ++j) { float v = gr[j]; s += v; q += v * v; }
      float mu  = s * (1.0f / 16.0f);
      float var = q * (1.0f / 16.0f) - mu * mu;
      float rs  = rsqrtf(var + 1e-5f);
      float acc = 0.f;
      #pragma unroll
      for (int j = 0; j < 16; ++j) {
        float v = (gr[j] - mu) * rs * ln_g_w[j] + ln_g_b[j];
        acc += gelu_exact(v) * w_g2[j];
      }
      gfeat[wv * 16 + lane] = acc + b_g2[0];
    }

    // ---------- phase 2: local branch (16x192)@(192x384), A held in registers ----------
    BfOp alh[6], alo[6];
    #pragma unroll
    for (int kt = 0; kt < 6; ++kt) {
      #pragma unroll
      for (int c2 = 0; c2 < 8; ++c2) {
        int c = 2 * c2;
        int K = kt * 32 + 16 * hh + c;
        split_pair(xrow[4 * K], xrow[4 * K + 4], alh[kt].u[c2], alo[kt].u[c2]);
      }
    }

    for (int nt = 0; nt < 24; ++nt) {
      __syncthreads();
      if (ws) {
        const char* hsrc = wsb + WS_LWHI_B + nt * 6144;
        const char* lsrc = wsb + WS_LWLO_B + nt * 6144;
        for (int i = tid * 16; i < 6144; i += NTHREADS * 16) {
          async_copy16(OFF_LW_HI + i, hsrc, i);
          async_copy16(OFF_LW_LO + i, lsrc, i);
        }
        async_wait0();
      } else {
        unsigned* dsthi = (unsigned*)(smem + OFF_LW_HI);
        unsigned* dstlo = (unsigned*)(smem + OFF_LW_LO);
        for (int i = tid; i < 16 * 96; i += NTHREADS) {
          int n = i / 96, kp = i - n * 96, k = 2 * kp;
          int src = k * OUT_ + nt * 16 + n;
          split_pair(w_l[src], w_l[src + OUT_], dsthi[i], dstlo[i]);
        }
      }
      __syncthreads();

      v8f cl = {};
      #pragma unroll
      for (int kt = 0; kt < 6; ++kt) {
        BfOp bh, bl;
        const uint4* bph = (const uint4*)(smem + OFF_LW_HI + (size_t)(ml * LOCALC_ + kt * 32 + 16 * hh) * 2);
        bh.q[0] = bph[0]; bh.q[1] = bph[1];
        const uint4* bpl = (const uint4*)(smem + OFF_LW_LO + (size_t)(ml * LOCALC_ + kt * 32 + 16 * hh) * 2);
        bl.q[0] = bpl[0]; bl.q[1] = bpl[1];
        cl = __builtin_amdgcn_wmma_f32_16x16x32_bf16(false, alh[kt].v, false, bh.v, (short)0, cl, false, false);
        cl = __builtin_amdgcn_wmma_f32_16x16x32_bf16(false, alo[kt].v, false, bh.v, (short)0, cl, false, false);
        cl = __builtin_amdgcn_wmma_f32_16x16x32_bf16(false, alh[kt].v, false, bl.v, (short)0, cl, false, false);
      }
      float blv = b_l[nt * 16 + ml];
      #pragma unroll
      for (int r = 0; r < 8; ++r)
        Ytile[(wv * 16 + r + 8 * hh) * OUT_ + nt * 16 + ml] = cl[r] + blv;
    }

    // ---------- phase 3: LN stats over 384 ----------
    __syncthreads();
    {
      float s = 0.f, q = 0.f;
      const float* yr = Ytile + (wv * 16 + ml) * OUT_ + 192 * hh;
      for (int j = 0; j < 48; ++j) {
        float4 y4 = *(const float4*)(yr + 4 * j);
        s += y4.x + y4.y + y4.z + y4.w;
        q += y4.x * y4.x + y4.y * y4.y + y4.z * y4.z + y4.w * y4.w;
      }
      s += __shfl_xor(s, 16, 32);
      q += __shfl_xor(q, 16, 32);
      float mu  = s * (1.0f / OUT_);
      float var = q * (1.0f / OUT_) - mu * mu;
      float rs  = rsqrtf(var + 1e-5f);
      if (lane < 16) { muA[wv * 16 + ml] = mu; rsA[wv * 16 + ml] = rs; }
    }
    __syncthreads();

    // ---------- phase 4: epilogue (coalesced) ----------
    for (int idx = tid; idx < ROWS_PER_WG * OUT_; idx += NTHREADS) {
      int rl = idx / OUT_, col = idx - rl * OUT_;
      int row = g * ROWS_PER_WG + rl;
      float y  = Ytile[rl * OUT_ + col];
      float lf = gelu_exact((y - muA[rl]) * rsA[rl] * ln_l_w[col] + ln_l_b[col]);
      float2 xp = *(const float2*)(x + (size_t)row * E_ + 2 * col);
      out[(size_t)row * OUT_ + col] = gfeat[rl] + 0.5f * (xp.x + xp.y) + lf;
    }
  }
}

extern "C" void kernel_launch(void* const* d_in, const int* in_sizes, int n_in,
                              void* d_out, int out_size, void* d_ws, size_t ws_size,
                              hipStream_t stream) {
  (void)in_sizes; (void)n_in; (void)out_size;
  const float* x      = (const float*)d_in[0];
  const float* w_g1   = (const float*)d_in[1];
  const float* b_g1   = (const float*)d_in[2];
  const float* ln_g_w = (const float*)d_in[3];
  const float* ln_g_b = (const float*)d_in[4];
  const float* w_g2   = (const float*)d_in[5];
  const float* b_g2   = (const float*)d_in[6];
  const float* w_l    = (const float*)d_in[7];
  const float* b_l    = (const float*)d_in[8];
  const float* ln_l_w = (const float*)d_in[9];
  const float* ln_l_b = (const float*)d_in[10];
  float* out = (float*)d_out;

  const bool use_ws = (d_ws != nullptr) && (ws_size >= (size_t)WS_BYTES);
  unsigned* ws = use_ws ? (unsigned*)d_ws : nullptr;
  if (use_ws)
    spectre_prep<<<dim3(168), dim3(256), 0, stream>>>(w_g1, w_l, ws);

  spectre_fused<<<dim3(NBLOCKS), dim3(NTHREADS), 0, stream>>>(
      x, w_g1, b_g1, ln_g_w, ln_g_b, w_g2, b_g2, w_l, b_l, ln_l_w, ln_l_b,
      (const unsigned*)ws, out);
}
